// MixtureOfExperts_76020921139217
// MI455X (gfx1250) — matrix operand order repaired and checked by reading, not compiled, not run on hardware
//
#include <hip/hip_runtime.h>
#include <hip/hip_bf16.h>

// ---------------------------------------------------------------------------
// MoE on gfx1250 (CDNA5, wave32):
//   prep:   fp32 -> bf16 conversion of x, W1^T, W2^T (one-time, halves traffic)
//   gate:   VALU gating + softmax + top-2 renorm
//   expert: dense per-expert MLP, bf16 WMMA w/ f32 accum,
//           double-buffered GLOBAL_LOAD_ASYNC_TO_LDS_B128 pipeline (ASYNCcnt)
//   combine: deterministic weighted sum
// ---------------------------------------------------------------------------

typedef __attribute__((ext_vector_type(16))) __bf16          v16bf;
typedef __attribute__((ext_vector_type(16))) unsigned short  v16us;
typedef __attribute__((ext_vector_type(8)))  unsigned short  v8us;
typedef __attribute__((ext_vector_type(4)))  unsigned short  v4us;
typedef __attribute__((ext_vector_type(8)))  float           v8f;

#define B_TOK 32768
#define DIM   512
#define HID   128
#define H2    64
#define NEXP  16
#define GHID  64
#define BT    128   // token tile per block
#define KC    64    // K chunk staged in LDS
#define NKC   (DIM / KC)   // 8 chunks (even, so W2 prefetch lands in buffer 0)

// float -> bf16 bits, round-to-nearest-even
__device__ __forceinline__ unsigned short f2bf(float f) {
  unsigned u = __float_as_uint(f);
  u += 0x7FFFu + ((u >> 16) & 1u);
  return (unsigned short)(u >> 16);
}

// CDNA5 async copy: 16 bytes global -> LDS, tracked by ASYNCcnt.
__device__ __forceinline__ void async_copy_b128(unsigned lds_off, const void* gptr) {
  asm volatile("global_load_async_to_lds_b128 %0, %1, off"
               :: "v"(lds_off), "v"(gptr)
               : "memory");
}
// Wait until at most N async ops remain outstanding (in-order completion).
template <int N>
__device__ __forceinline__ void wait_async_le() {
  asm volatile("s_wait_asynccnt %0" :: "i"(N) : "memory");
}

// A-fragment (16x32 bf16, ISA 7.12.2): lane half h -> K = {8h..8h+7, 16+8h..16+8h+7}
__device__ __forceinline__ v16bf load_frag_a(const unsigned short* p, int half) {
  const int ak = half * 8;
  v8us lo = *(const v8us*)(p + ak);
  v8us hi = *(const v8us*)(p + 16 + ak);
  v16us u;
#pragma unroll
  for (int i = 0; i < 8; ++i) { u[i] = lo[i]; u[i + 8] = hi[i]; }
  return __builtin_bit_cast(v16bf, u);
}

// B-fragment (32x16 bf16, K-contiguous rows in LDS): lane half h -> K = {8h..8h+7, 16+8h..16+8h+7}
__device__ __forceinline__ v16bf load_frag_b(const unsigned short* p, int half) {
  const int bk = half * 8;
  v8us lo = *(const v8us*)(p + bk);
  v8us hi = *(const v8us*)(p + 16 + bk);
  v16us u;
#pragma unroll
  for (int i = 0; i < 8; ++i) { u[i] = lo[i]; u[i + 8] = hi[i]; }
  return __builtin_bit_cast(v16bf, u);
}

// ---------------------------------------------------------------------------
// Prep kernels: one-time fp32 -> bf16 (x) and transpose+convert (W1, W2)
// ---------------------------------------------------------------------------
__global__ __launch_bounds__(256) void convert_x_kernel(
    const float* __restrict__ x, unsigned short* __restrict__ xbf)
{
  size_t i = ((size_t)blockIdx.x * 256 + threadIdx.x) * 4;
  float4 v = *(const float4*)(x + i);
  v4us o;
  o[0] = f2bf(v.x); o[1] = f2bf(v.y); o[2] = f2bf(v.z); o[3] = f2bf(v.w);
  *(v4us*)(xbf + i) = o;
}

// w1t[e][h][d] = bf16(W1[e][d][h])   (K-contiguous rows for B fragments)
__global__ __launch_bounds__(256) void transpose_w1_kernel(
    const float* __restrict__ W1, unsigned short* __restrict__ w1t)
{
  size_t i = (size_t)blockIdx.x * 256 + threadIdx.x;   // over E*H*D
  int d = (int)(i % DIM);
  size_t eh = i / DIM;
  int h = (int)(eh % HID);
  int e = (int)(eh / HID);
  w1t[i] = f2bf(W1[((size_t)e * DIM + d) * HID + h]);
}

// w2t[e][c][h] = bf16(W2[e][h][c])
__global__ __launch_bounds__(256) void transpose_w2_kernel(
    const float* __restrict__ W2, unsigned short* __restrict__ w2t)
{
  size_t i = (size_t)blockIdx.x * 256 + threadIdx.x;   // over E*H2*H
  int h = (int)(i % HID);
  size_t ec = i / HID;
  int c = (int)(ec % H2);
  int e = (int)(ec / H2);
  w2t[i] = f2bf(W2[((size_t)e * HID + h) * H2 + c]);
}

// ---------------------------------------------------------------------------
// Gating network.  One wave (32 lanes) per token.
// ---------------------------------------------------------------------------
__global__ __launch_bounds__(256) void moe_gate_kernel(
    const float* __restrict__ x,   const float* __restrict__ Wg1,
    const float* __restrict__ bg1, const float* __restrict__ Wg2,
    const float* __restrict__ bg2, float* __restrict__ gates)
{
  __shared__ float gh[8][GHID];
  __shared__ float gl[8][NEXP];
  const int w    = threadIdx.x >> 5;
  const int lane = threadIdx.x & 31;
  const int b    = blockIdx.x * 8 + w;

  float a0 = 0.f, a1 = 0.f;
  for (int d = 0; d < DIM; ++d) {
    float xv = x[(size_t)b * DIM + d];
    a0 = fmaf(xv, Wg1[d * GHID + lane],      a0);
    a1 = fmaf(xv, Wg1[d * GHID + lane + 32], a1);
  }
  a0 += bg1[lane];      if (a0 < 0.f) a0 = 0.f;
  a1 += bg1[lane + 32]; if (a1 < 0.f) a1 = 0.f;
  gh[w][lane]      = a0;
  gh[w][lane + 32] = a1;
  __syncthreads();

  if (lane < NEXP) {
    float s = bg2[lane];
    for (int g = 0; g < GHID; ++g) s = fmaf(gh[w][g], Wg2[g * NEXP + lane], s);
    gl[w][lane] = s;
  }
  __syncthreads();

  if (lane == 0) {
    float mx = gl[w][0];
    for (int e = 1; e < NEXP; ++e) mx = fmaxf(mx, gl[w][e]);
    float p[NEXP], sum = 0.f;
    for (int e = 0; e < NEXP; ++e) { p[e] = __expf(gl[w][e] - mx); sum += p[e]; }
    float inv = 1.f / sum;
    for (int e = 0; e < NEXP; ++e) p[e] *= inv;
    int t1 = 0;
    for (int e = 1; e < NEXP; ++e) if (p[e] > p[t1]) t1 = e;
    int t2 = (t1 == 0) ? 1 : 0;
    for (int e = 0; e < NEXP; ++e) if (e != t1 && p[e] > p[t2]) t2 = e;
    float den = 1.f / (p[t1] + p[t2] + 1e-10f);
    for (int e = 0; e < NEXP; ++e)
      gates[(size_t)b * NEXP + e] = (e == t1 || e == t2) ? p[e] * den : 0.f;
  }
}

// ---------------------------------------------------------------------------
// Dense expert MLPs via bf16 WMMA.  grid = (B/128, E), 8 waves per block.
// Double-buffered async global->LDS pipeline: copies for chunk k+1 overlap
// WMMA on chunk k (ASYNCcnt in-order completion => partial waits).
// ---------------------------------------------------------------------------
__global__ __launch_bounds__(256) void moe_expert_kernel(
    const unsigned short* __restrict__ xbf,  // [B][D]    bf16
    const unsigned short* __restrict__ w1t,  // [E][H][D] bf16 (transposed)
    const float* __restrict__ b1,
    const unsigned short* __restrict__ w2t,  // [E][H2][H] bf16 (transposed)
    const float* __restrict__ b2,
    const float* __restrict__ W3, const float* __restrict__ b3,
    float* __restrict__ outE)
{
  __shared__ alignas(16) unsigned short lds_x[2][BT * KC];    // x chunks   (buf0 reused: W2^T [64][128])
  __shared__ alignas(16) unsigned short lds_w[2][HID * KC];   // W1^T chunks
  __shared__ alignas(16) unsigned short lds_h1[BT * HID];     // h1 bf16 [128][128]

  const int e    = blockIdx.y;
  const int b0   = blockIdx.x * BT;
  const int tid  = threadIdx.x;
  const int w    = tid >> 5;
  const int lane = tid & 31;
  const int m    = lane & 15;
  const int n    = lane & 15;
  const int half = lane >> 4;

  const unsigned short* w1te = w1t + (size_t)e * HID * DIM;

  // issue one chunk's async copies: 8 x b128 per thread (4 for x, 4 for W1^T)
  auto issue_chunk = [&](int buf, int kc) {
    const unsigned xbase = (unsigned)(uintptr_t)&lds_x[buf][0];
    const unsigned wbase = (unsigned)(uintptr_t)&lds_w[buf][0];
#pragma unroll
    for (int i = tid; i < BT * KC / 8; i += 256) {
      int r = i >> 3, p = i & 7;
      async_copy_b128(xbase + (unsigned)i * 16,
                      xbf + (size_t)(b0 + r) * DIM + kc * KC + p * 8);
    }
#pragma unroll
    for (int i = tid; i < HID * KC / 8; i += 256) {
      int h = i >> 3, p = i & 7;
      async_copy_b128(wbase + (unsigned)i * 16,
                      w1te + (size_t)h * DIM + kc * KC + p * 8);
    }
  };

  // ---- layer 1: h1 = relu(x @ W1[e] + b1[e]) --------------------------------
  v8f acc[8] = {};
  issue_chunk(0, 0);                       // prologue
  for (int kc = 0; kc < NKC; ++kc) {
    const int cur = kc & 1;
    __syncthreads();                       // buffer cur^1 free (readers done)
    if (kc + 1 < NKC) {
      issue_chunk(cur ^ 1, kc + 1);        // +8 per thread
      wait_async_le<8>();                  // chunk kc landed; kc+1 in flight
    } else {
      // final chunk: stream W2^T [64][128] into lds_x[0] underneath the tail
      const unsigned short* w2te = w2t + (size_t)e * H2 * HID;
      const unsigned xbase = (unsigned)(uintptr_t)&lds_x[cur ^ 1][0];
#pragma unroll
      for (int i = tid; i < H2 * HID / 8; i += 256)
        async_copy_b128(xbase + (unsigned)i * 16, w2te + (size_t)i * 8);
      wait_async_le<4>();                  // chunk kc landed; W2 (4) in flight
    }
    __syncthreads();                       // all waves' chunk kc copies visible
#pragma unroll
    for (int ks = 0; ks < KC; ks += 32) {
      v16bf a = load_frag_a(&lds_x[cur][(w * 16 + m) * KC + ks], half);
      v16bf bf[8];
#pragma unroll
      for (int nt = 0; nt < 8; ++nt)
        bf[nt] = load_frag_b(&lds_w[cur][(nt * 16 + n) * KC + ks], half);
#pragma unroll
      for (int nt = 0; nt < 8; ++nt)
        acc[nt] = __builtin_amdgcn_wmma_f32_16x16x32_bf16(
            false, a, false, bf[nt], (short)0, acc[nt], false, false);
    }
  }
  // bias + relu -> bf16 h1 in LDS (disjoint from in-flight W2 copies)
#pragma unroll
  for (int nt = 0; nt < 8; ++nt) {
    float bv = b1[e * HID + nt * 16 + n];
#pragma unroll
    for (int r = 0; r < 8; ++r) {
      float v = acc[nt][r] + bv;
      if (v < 0.f) v = 0.f;
      lds_h1[(w * 16 + 8 * half + r) * HID + nt * 16 + n] = f2bf(v);
    }
  }
  wait_async_le<0>();                      // W2^T copies complete
  __syncthreads();

  // ---- layer 2: h2 = relu(h1 @ W2[e] + b2[e]) -------------------------------
  v8f acc2[4] = {};
#pragma unroll
  for (int ks = 0; ks < HID; ks += 32) {
    v16bf a = load_frag_a(&lds_h1[(w * 16 + m) * HID + ks], half);
    v16bf bf[4];
#pragma unroll
    for (int nt = 0; nt < 4; ++nt)
      bf[nt] = load_frag_b(&lds_x[0][(nt * 16 + n) * HID + ks], half);
#pragma unroll
    for (int nt = 0; nt < 4; ++nt)
      acc2[nt] = __builtin_amdgcn_wmma_f32_16x16x32_bf16(
          false, a, false, bf[nt], (short)0, acc2[nt], false, false);
  }

  // ---- layer 3: out = sigmoid(h2 @ W3[e] + b3[e]) ---------------------------
  float pr[8] = {0.f, 0.f, 0.f, 0.f, 0.f, 0.f, 0.f, 0.f};
#pragma unroll
  for (int nt = 0; nt < 4; ++nt) {
    float b2v = b2[e * H2 + nt * 16 + n];
    float w3v = W3[(size_t)e * H2 + nt * 16 + n];
#pragma unroll
    for (int r = 0; r < 8; ++r) {
      float h2v = acc2[nt][r] + b2v;
      if (h2v < 0.f) h2v = 0.f;
      pr[r] = fmaf(h2v, w3v, pr[r]);
    }
  }
#pragma unroll
  for (int off = 1; off < 16; off <<= 1) {
#pragma unroll
    for (int r = 0; r < 8; ++r) pr[r] += __shfl_xor(pr[r], off, 32);
  }
  if (n == 0) {
    float b3v = b3[e];
#pragma unroll
    for (int r = 0; r < 8; ++r) {
      float s = 1.f / (1.f + __expf(-(pr[r] + b3v)));
      outE[(size_t)e * B_TOK + b0 + w * 16 + 8 * half + r] = s;
    }
  }
}

// ---------------------------------------------------------------------------
// Deterministic combine:  pred[b] = sum_e gates[b,e] * outE[e,b]
// ---------------------------------------------------------------------------
__global__ __launch_bounds__(256) void moe_combine_kernel(
    const float* __restrict__ gates, const float* __restrict__ outE,
    float* __restrict__ pred)
{
  int b = blockIdx.x * blockDim.x + threadIdx.x;
  if (b >= B_TOK) return;
  float s = 0.f;
#pragma unroll
  for (int e = 0; e < NEXP; ++e)
    s = fmaf(gates[(size_t)b * NEXP + e], outE[(size_t)e * B_TOK + b], s);
  pred[b] = s;
}

// ---------------------------------------------------------------------------
extern "C" void kernel_launch(void* const* d_in, const int* in_sizes, int n_in,
                              void* d_out, int out_size, void* d_ws, size_t ws_size,
                              hipStream_t stream) {
  const float* x   = (const float*)d_in[0];
  const float* Wg1 = (const float*)d_in[1];
  const float* bg1 = (const float*)d_in[2];
  const float* Wg2 = (const float*)d_in[3];
  const float* bg2 = (const float*)d_in[4];
  const float* W1  = (const float*)d_in[5];
  const float* b1  = (const float*)d_in[6];
  const float* W2  = (const float*)d_in[7];
  const float* b2  = (const float*)d_in[8];
  const float* W3  = (const float*)d_in[9];
  const float* b3  = (const float*)d_in[10];

  // workspace layout
  float*          ws_gates = (float*)d_ws;                        // [B,E]    2 MB
  float*          ws_outE  = ws_gates + (size_t)B_TOK * NEXP;     // [E,B]    2 MB
  unsigned short* ws_xbf   = (unsigned short*)(ws_outE + (size_t)NEXP * B_TOK);  // 32 MB
  unsigned short* ws_w1t   = ws_xbf + (size_t)B_TOK * DIM;        // [E,H,D]  2 MB
  unsigned short* ws_w2t   = ws_w1t + (size_t)NEXP * HID * DIM;   // [E,H2,H] 256 KB

  // prep: one-time bf16 conversion / transposes
  convert_x_kernel<<<(B_TOK * DIM) / (256 * 4), 256, 0, stream>>>(x, ws_xbf);
  transpose_w1_kernel<<<(NEXP * HID * DIM) / 256, 256, 0, stream>>>(W1, ws_w1t);
  transpose_w2_kernel<<<(NEXP * H2 * HID) / 256, 256, 0, stream>>>(W2, ws_w2t);

  moe_gate_kernel<<<B_TOK / 8, 256, 0, stream>>>(x, Wg1, bg1, Wg2, bg2, ws_gates);

  dim3 grid(B_TOK / BT, NEXP);
  moe_expert_kernel<<<grid, 256, 0, stream>>>(ws_xbf, ws_w1t, b1, ws_w2t, b2, W3, b3, ws_outE);

  moe_combine_kernel<<<B_TOK / 256, 256, 0, stream>>>(ws_gates, ws_outE, (float*)d_out);
}
